// MessagePassingConvolution_63934883168899
// MI455X (gfx1250) — matrix operand order repaired and checked by reading, not compiled, not run on hardware
//
#include <hip/hip_runtime.h>

// MessagePassingConvolution for MI455X (gfx1250), wave32 + WMMA.
// MLP layers 2 and 3 run on v_wmma_f32_16x16x32_f16 (f16 in, f32 accum).

typedef __attribute__((ext_vector_type(16))) _Float16 v16h;
typedef __attribute__((ext_vector_type(8)))  _Float16 v8h;
typedef __attribute__((ext_vector_type(8)))  float    v8f;

#define ACT_C     1.6790590f      // 1/sqrt(E[silu(Z)^2]), Z~N(0,1)
#define SQRT2_F   1.41421356237f
#define SQRT3_F   1.73205080757f
#define PI_F      3.14159265358979f
#define INV_SQRT8 0.3535533906f   // layer-1 fan-in scale
#define L3_SCALE  0.03125f        // (1/sqrt(64)) * (1/sqrt(16))

// ---- WMMA fragment loaders (contiguous k = 16h + s order, same in A and B) -------------------------
// A (16x32 f16), row-major LDS tile with row stride 64 halves:
//   lanes 0-15 : row = lane,    K = kk*32 + 0..15
//   lanes 16-31: row = lane-16, K = kk*32 + 16..31
__device__ __forceinline__ v16h loadA(const _Float16* tileBase, int lane, int kk) {
    const int row = lane & 15;
    const int grp = (lane >> 4) << 4;                 // 0 or 16
    const _Float16* p = tileBase + row * 64 + kk * 32 + grp;
    v8h g0 = *(const v8h*)(p);                        // 8 contiguous halves (b128)
    v8h g1 = *(const v8h*)(p + 8);                    // 8 contiguous halves (b128)
    return __builtin_shufflevector(g0, g1, 0,1,2,3,4,5,6,7,8,9,10,11,12,13,14,15);
}

// B (32x16 f16) from a transposed weight table wT[N][64] in LDS:
//   lanes 0-15 : N = n0+lane,    K = kk*32 + 0..15
//   lanes 16-31: N = n0+lane-16, K = kk*32 + 16..31
__device__ __forceinline__ v16h loadB(const _Float16* wT, int lane, int n0, int kk) {
    const int n  = n0 + (lane & 15);
    const int kb = kk * 32 + ((lane >> 4) << 4);      // 0 or 16
    return *(const v16h*)(wT + n * 64 + kb);          // 32B contiguous
}

// swish with hardware v_rcp_f32 (avoids the precise-division refinement chain)
__device__ __forceinline__ float swish_act(float x) {
    return ACT_C * x * __builtin_amdgcn_rcpf(1.f + __expf(-x));
}

// ---- zero the output (atomics accumulate into it) ---------------------------
__global__ void zero_out_kernel(float* __restrict__ out, int n) {
    int i = blockIdx.x * blockDim.x + threadIdx.x;
    if (i < n) out[i] = 0.f;
}

// ---- main fused kernel: 128 threads (4 waves), 64 edges per block -----------
__global__ __launch_bounds__(128)
void mpconv_kernel(const float* __restrict__ vectors,       // [E][3]
                   const float* __restrict__ node_scalars,  // [N][64]
                   const float* __restrict__ node_vectors,  // [N][64][3]
                   const int*   __restrict__ senders,       // [E]
                   const int*   __restrict__ receivers,     // [E]
                   const float* __restrict__ W1,            // [8][64]
                   const float* __restrict__ W2,            // [64][64]
                   const float* __restrict__ W3,            // [64][256]
                   float* __restrict__ out,                 // [N][512]
                   int nEdges)
{
    __shared__ __align__(32) float    sW1 [8 * 64];      //  2 KB
    __shared__ __align__(32) _Float16 sW2T[64 * 64];     //  8 KB (W2 transposed, f16)
    __shared__ __align__(32) _Float16 sW3T[256 * 64];    // 32 KB (W3 transposed, f16)
    __shared__ __align__(32) _Float16 sH1 [64 * 64];     //  8 KB
    __shared__ __align__(32) _Float16 sH2 [64 * 64];     //  8 KB
    __shared__ float sY[64 * 3];                         // v/len per edge

    const int t = threadIdx.x;
    const int blockE0 = blockIdx.x * 64;

    // --- stage weights into LDS (W2/W3 transposed to [N][K] f16) ---
    for (int i = t; i < 8 * 64; i += 128) sW1[i] = W1[i];
    for (int i = t; i < 64 * 64; i += 128) {
        int k = i >> 6, n = i & 63;
        sW2T[n * 64 + k] = (_Float16)W2[i];
    }
    for (int i = t; i < 64 * 256; i += 128) {
        int k = i >> 8, n = i & 255;
        sW3T[n * 64 + k] = (_Float16)W3[i];
    }
    __syncthreads();

    // --- phase 1: radial basis + layer 1 (8->64) in VALU, 2 threads/edge ---
    {
        const int eLoc = t >> 1;
        const int half = t & 1;
        const int edge = min(blockE0 + eLoc, nEdges - 1);
        const float vx = vectors[edge * 3 + 0];
        const float vy = vectors[edge * 3 + 1];
        const float vz = vectors[edge * 3 + 2];
        const float len = sqrtf(vx * vx + vy * vy + vz * vz);
        const float invl = (len > 0.f) ? __builtin_amdgcn_rcpf(len) : 0.f;
        if (half == 0) {
            sY[eLoc * 3 + 0] = vx * invl;   // v/len  (Y1 = sqrt(3)*v/len)
            sY[eLoc * 3 + 1] = vy * invl;
            sY[eLoc * 3 + 2] = vz * invl;
        }
        float env = 0.f;
        if (len < 1.f) {
            float x2 = len * len, x4 = x2 * x2, x6 = x4 * x2;
            float x7 = x6 * len, x8 = x7 * len;
            env = 1.f - 28.f * x6 + 48.f * x7 - 21.f * x8;
        }
        float rad[8];
        #pragma unroll
        for (int k = 1; k <= 8; ++k) {
            float b = (len == 0.f) ? (k * PI_F) : (__sinf(PI_F * k * len) * invl);
            rad[k - 1] = SQRT2_F * b * env;
        }
        #pragma unroll 4
        for (int j = 0; j < 32; ++j) {
            const int n = half * 32 + j;
            float acc = 0.f;
            #pragma unroll
            for (int k = 0; k < 8; ++k) acc += rad[k] * sW1[k * 64 + n];
            sH1[eLoc * 64 + n] = (_Float16)swish_act(acc * INV_SQRT8);
        }
    }
    __syncthreads();

    const int lane  = t & 31;
    const int e0    = (t >> 5) * 16;        // this wave's 16-edge tile
    const int rOff  = (lane >> 4) << 3;     // C/D layout: upper lanes hold M+8
    const int nLane = lane & 15;

    // --- layer 2: h2 = act(h1 @ W2 / 8), WMMA 16x16x32 f16 ---
    {
        const v16h a0 = loadA(sH1 + e0 * 64, lane, 0);   // A independent of nt: hoist
        const v16h a1 = loadA(sH1 + e0 * 64, lane, 1);
        #pragma unroll
        for (int nt = 0; nt < 4; ++nt) {
            v8f c = {0.f, 0.f, 0.f, 0.f, 0.f, 0.f, 0.f, 0.f};
            c = __builtin_amdgcn_wmma_f32_16x16x32_f16(
                    false, a0, false, loadB(sW2T, lane, nt * 16, 0), (short)0, c, false, false);
            c = __builtin_amdgcn_wmma_f32_16x16x32_f16(
                    false, a1, false, loadB(sW2T, lane, nt * 16, 1), (short)0, c, false, false);
            const int nCol = nt * 16 + nLane;
            #pragma unroll
            for (int r = 0; r < 8; ++r) {
                float x = c[r] * 0.125f;    // 1/sqrt(64)
                sH2[(e0 + r + rOff) * 64 + nCol] = (_Float16)swish_act(x);
            }
        }
    }
    __syncthreads();

    // --- layer 3 + gather/scatter: mix = h2 @ W3 / 8, then message scatter ---
    {
        const v16h a0 = loadA(sH2 + e0 * 64, lane, 0);   // hoisted A fragments
        const v16h a1 = loadA(sH2 + e0 * 64, lane, 1);
        for (int nt = 0; nt < 16; ++nt) {
            v8f c = {0.f, 0.f, 0.f, 0.f, 0.f, 0.f, 0.f, 0.f};
            c = __builtin_amdgcn_wmma_f32_16x16x32_f16(
                    false, a0, false, loadB(sW3T, lane, nt * 16, 0), (short)0, c, false, false);
            c = __builtin_amdgcn_wmma_f32_16x16x32_f16(
                    false, a1, false, loadB(sW3T, lane, nt * 16, 1), (short)0, c, false, false);
            const int col = nt * 16 + nLane;       // mix channel 0..255
            #pragma unroll
            for (int r = 0; r < 8; ++r) {
                const int e    = e0 + r + rOff;    // edge within block
                const int edge = blockE0 + e;
                if (edge >= nEdges) continue;
                const int snd = senders[edge];
                const int rcv = receivers[edge];
                const float val = c[r] * L3_SCALE; // /sqrt(64) * 1/sqrt(avg_neigh)
                float* outN = out + (size_t)rcv * 512;
                if (col < 64) {                    // s part 1: m0 * mix
                    atomicAdd(outN + col, node_scalars[(size_t)snd * 64 + col] * val);
                } else if (col < 128) {            // s part 2: tp0 * mix = (m1 . v/len)*mix
                    const int u = col - 64;
                    const float* m1 = node_vectors + (size_t)snd * 192 + u * 3;
                    float tp0 = m1[0] * sY[e * 3 + 0] + m1[1] * sY[e * 3 + 1]
                              + m1[2] * sY[e * 3 + 2];
                    atomicAdd(outN + col, tp0 * val);
                } else if (col < 192) {            // v part 1: m1 * mix
                    const int u = col - 128;
                    const float* m1 = node_vectors + (size_t)snd * 192 + u * 3;
                    float* o = outN + 128 + u * 3;
                    atomicAdd(o + 0, m1[0] * val);
                    atomicAdd(o + 1, m1[1] * val);
                    atomicAdd(o + 2, m1[2] * val);
                } else {                           // v part 2: m0 * Y1 * mix
                    const int u = col - 192;
                    const float f = node_scalars[(size_t)snd * 64 + u] * val * SQRT3_F;
                    float* o = outN + 128 + (64 + u) * 3;
                    atomicAdd(o + 0, f * sY[e * 3 + 0]);
                    atomicAdd(o + 1, f * sY[e * 3 + 1]);
                    atomicAdd(o + 2, f * sY[e * 3 + 2]);
                }
            }
        }
    }
}

extern "C" void kernel_launch(void* const* d_in, const int* in_sizes, int n_in,
                              void* d_out, int out_size, void* d_ws, size_t ws_size,
                              hipStream_t stream) {
    const float* vectors      = (const float*)d_in[0];
    const float* node_scalars = (const float*)d_in[1];
    const float* node_vectors = (const float*)d_in[2];
    const int*   senders      = (const int*)d_in[3];
    const int*   receivers    = (const int*)d_in[4];
    const float* W1           = (const float*)d_in[5];
    const float* W2           = (const float*)d_in[6];
    const float* W3           = (const float*)d_in[7];
    float* out = (float*)d_out;
    (void)n_in; (void)d_ws; (void)ws_size;

    const int nEdges = in_sizes[0] / 3;

    // zero the accumulation target every call (harness poisons once, atomics add)
    zero_out_kernel<<<(out_size + 255) / 256, 256, 0, stream>>>(out, out_size);

    const int nBlocks = (nEdges + 63) / 64;
    mpconv_kernel<<<nBlocks, 128, 0, stream>>>(vectors, node_scalars, node_vectors,
                                               senders, receivers, W1, W2, W3,
                                               out, nEdges);
}
